// TransformerBlock_10883447128153
// MI455X (gfx1250) — matrix operand order repaired and checked by reading, not compiled, not run on hardware
//
#include <hip/hip_runtime.h>
#include <hip/hip_bf16.h>

// ---------------------------------------------------------------------------
// Types for gfx1250 WMMA (codegen-confirmed builtin signature):
//   v8f = __builtin_amdgcn_wmma_f32_16x16x32_f16(bool, v16h, bool, v16h,
//                                                short, v8f, bool, bool)
// ---------------------------------------------------------------------------
typedef _Float16 half_t;
typedef __attribute__((ext_vector_type(16))) _Float16 v16h;
typedef __attribute__((ext_vector_type(8)))  float    v8f;

#define Bn     4
#define Cc     128
#define HWpix  65536          // 256*256
#define NHEADS 8
#define NFREQ  33024          // 256*129 (logical)
#define FLD    136            // padded freq leading dim (272B rows, 16B aligned)
#define FPI    34816          // 256*136 padded freq elems per image (=272*128)
#define THETA  0.7f

// ===========================================================================
// Generic f16-in / f32-accumulate WMMA GEMM:  C = alpha*A*B (+C) (+bias)(+res)
//   A: [M,K] row-major f16   (per-batch stride sA, 0 => shared)
//   B: [K,N] row-major f16
//   C: f32 out (optional), C16: f16 shadow copy (optional)
//   act: 0 none, 1 leaky_relu(0.1), 2 abs
// Fast path (full tile): A staged via gfx1250 async DMA global->LDS,
// B staged via aligned b128 loads + b64 LDS stores into transposed image.
// ===========================================================================
#define TM 128
#define TN 128
#define TK 32

__global__ __launch_bounds__(256)
void gemm_f16_kernel(const half_t* __restrict__ A, long long sA, int lda,
                     const half_t* __restrict__ Bm, long long sB, int ldb,
                     float* C, long long sC, int ldc,
                     half_t* C16, long long sC16, int ldc16,
                     const float* __restrict__ bias,
                     const float* __restrict__ Res, long long sR, int ldr,
                     int M, int K, int N, float alpha, int accumulate, int act)
{
    __shared__ __align__(16) half_t As [TM][TK + 8];   // 80B rows, 16B aligned
    __shared__ __align__(16) half_t BsT[TN][TK + 8];   // B stored transposed

    const int zb    = blockIdx.z;
    const half_t* Ab = A  + (long long)zb * sA;
    const half_t* Bb = Bm + (long long)zb * sB;

    const int nBase = blockIdx.x * TN;
    const int mBase = blockIdx.y * TM;
    const int tid   = threadIdx.x;
    const int lane  = tid & 31;
    const int wave  = tid >> 5;
    const int wm    = wave & 3;    // 4 wave-rows  * 32 M each
    const int wn    = wave >> 2;   // 2 wave-cols  * 64 N each

    v8f acc[2][4];
    for (int i = 0; i < 2; ++i)
        for (int j = 0; j < 4; ++j)
            for (int e = 0; e < 8; ++e) acc[i][j][e] = 0.0f;

    const bool fullN = (nBase + TN) <= N;   // M tiles are always full here
    const int kTiles = (K + TK - 1) / TK;
    for (int kt = 0; kt < kTiles; ++kt) {
        const int k0 = kt * TK;
        const bool full = fullN && (k0 + TK) <= K;
        if (full) {
            if (tid < 128) {
                // ---- B tile: 8n x 4k register block, aligned b128 loads ----
                const int seg = tid & 15;        // 16 segments of 8 n
                const int kq  = tid >> 4;        // 8 groups of 4 k rows
                const int n0  = seg * 8;
                const half_t* src =
                    Bb + (long long)(k0 + kq * 4) * ldb + nBase + n0;
                half_t tmp[4][8];
#pragma unroll
                for (int r = 0; r < 4; ++r)
                    __builtin_memcpy(tmp[r], src + (long long)r * ldb, 16);
#pragma unroll
                for (int nn = 0; nn < 8; ++nn) {
                    half_t pk[4] = {tmp[0][nn], tmp[1][nn],
                                    tmp[2][nn], tmp[3][nn]};
                    __builtin_memcpy(&BsT[n0 + nn][kq * 4], pk, 8);
                }
            } else {
                // ---- A tile: one 64B row/thread via async DMA to LDS ------
                const int m = tid - 128;
                unsigned lbase = (unsigned)(uintptr_t)(&As[m][0]);
                const half_t* gp = Ab + (long long)(mBase + m) * lda + k0;
#pragma unroll
                for (int c4 = 0; c4 < 4; ++c4)
                    asm volatile(
                        "global_load_async_to_lds_b128 %0, %1, off"
                        :: "v"(lbase + (unsigned)(c4 * 16)), "v"(gp + c4 * 8)
                        : "memory");
            }
            if (kt + 1 < kTiles) {
                int gn = nBase + (tid & 127);
                int gk = k0 + TK + (tid >> 7);
                if (gk < K)
                    __builtin_prefetch(&Bb[(long long)gk * ldb + gn], 0, 1);
            }
            asm volatile("s_wait_asynccnt 0x0" ::: "memory");
        } else {
            // ---- guarded edge-tile staging (rare) ----
            for (int i = tid; i < TM * TK; i += 256) {
                int r  = i >> 5;
                int kk = i & 31;
                int gm = mBase + r, gk = k0 + kk;
                As[r][kk] = (gm < M && gk < K)
                              ? Ab[(long long)gm * lda + gk] : (half_t)0.0f;
            }
            for (int i = tid; i < TN * TK; i += 256) {
                int c  = i & (TN - 1);
                int kk = i >> 7;
                int gn = nBase + c, gk = k0 + kk;
                BsT[c][kk] = (gn < N && gk < K)
                               ? Bb[(long long)gk * ldb + gn] : (half_t)0.0f;
            }
        }
        __syncthreads();

        // ---- fragments: contiguous k = 16*(lane/16) + s in A and B ----
        const int lr   = lane & 15;
        const int koff = (lane >> 4) << 3;   // unused (0 or 8)
        const int kb   = (lane >> 4) << 4;   // A and B: 0 or 16
        v16h af[2], bf[4];
        for (int i = 0; i < 2; ++i) {
            int row = wm * 32 + i * 16 + lr;
            __builtin_memcpy(&af[i], &As[row][kb], 32);
        }
        for (int j = 0; j < 4; ++j) {
            int col = wn * 64 + j * 16 + lr;
            __builtin_memcpy(&bf[j], &BsT[col][kb], 32);
        }
        for (int i = 0; i < 2; ++i)
            for (int j = 0; j < 4; ++j)
                acc[i][j] = __builtin_amdgcn_wmma_f32_16x16x32_f16(
                    false, af[i], false, bf[j], (short)0, acc[i][j],
                    false, false);
        __syncthreads();
    }

    // ---- epilogue: C/D layout (lane = N col, vgpr r = M row, hi-lanes +8) --
    float*        Cb   = C   ? C   + (long long)zb * sC   : nullptr;
    half_t*       C16b = C16 ? C16 + (long long)zb * sC16 : nullptr;
    const float*  Rb   = Res ? Res + (long long)zb * sR   : nullptr;
    const int rbase = (lane >> 4) << 3;
    const int cl    = lane & 15;
    for (int i = 0; i < 2; ++i)
        for (int j = 0; j < 4; ++j) {
            int gm0 = mBase + wm * 32 + i * 16;
            int gn  = nBase + wn * 64 + j * 16 + cl;
            if (gn >= N) continue;
            for (int r = 0; r < 8; ++r) {
                int gm = gm0 + rbase + r;
                if (gm >= M) continue;
                float v = alpha * acc[i][j][r];
                if (accumulate && Cb) v += Cb[(long long)gm * ldc + gn];
                if (bias) v += bias[gm];
                if (Rb)   v += Rb[(long long)gm * ldr + gn];
                if (act == 1)      v = v > 0.0f ? v : 0.1f * v;
                else if (act == 2) v = fabsf(v);
                if (Cb)   Cb  [(long long)gm * ldc   + gn] = v;
                if (C16b) C16b[(long long)gm * ldc16 + gn] = (half_t)v;
            }
        }
}

// ===========================================================================
// Elementwise / small kernels
// ===========================================================================
__global__ void cvt_f32_f16_kernel(const float* __restrict__ in,
                                   half_t* __restrict__ out,
                                   long long n, int doAbs)
{
    long long i = (long long)blockIdx.x * 256 + threadIdx.x;
    if (i >= n) return;
    float v = in[i];
    if (doAbs) v = fabsf(v);
    out[i] = (half_t)v;
}

// per-pixel LayerNorm over 128 channels -> f16
__global__ void layernorm_kernel(const float* __restrict__ x,
                                 const float* __restrict__ w,
                                 const float* __restrict__ b2,
                                 half_t* __restrict__ out)
{
    long long p = (long long)blockIdx.x * 256 + threadIdx.x;
    if (p >= (long long)Bn * HWpix) return;
    long long b = p >> 16, hw = p & 65535;
    const float* xb = x + (b * Cc) * (long long)HWpix + hw;
    float s = 0.f, ss = 0.f;
    for (int c = 0; c < Cc; ++c) { float v = xb[(long long)c * HWpix]; s += v; ss += v * v; }
    float mu = s / Cc, var = ss / Cc - mu * mu;
    float inv = rsqrtf(var + 1e-5f);
    half_t* ob = out + (b * Cc) * (long long)HWpix + hw;
    for (int c = 0; c < Cc; ++c) {
        float v = xb[(long long)c * HWpix];
        ob[(long long)c * HWpix] = (half_t)((v - mu) * inv * w[c] + b2[c]);
    }
}

// per-pixel LayerNorm over concat(x, frefuse) = 256 channels -> f16
__global__ void layernorm2_cat_kernel(const float* __restrict__ x1,
                                      const float* __restrict__ x2,
                                      const float* __restrict__ w,
                                      const float* __restrict__ b2,
                                      half_t* __restrict__ out)
{
    long long p = (long long)blockIdx.x * 256 + threadIdx.x;
    if (p >= (long long)Bn * HWpix) return;
    long long b = p >> 16, hw = p & 65535;
    const float* a1 = x1 + (b * Cc) * (long long)HWpix + hw;
    const float* a2 = x2 + (b * Cc) * (long long)HWpix + hw;
    float s = 0.f, ss = 0.f;
    for (int c = 0; c < Cc; ++c) { float v = a1[(long long)c * HWpix]; s += v; ss += v * v; }
    for (int c = 0; c < Cc; ++c) { float v = a2[(long long)c * HWpix]; s += v; ss += v * v; }
    float mu = s / 256.f, var = ss / 256.f - mu * mu;
    float inv = rsqrtf(var + 1e-5f);
    half_t* ob = out + (b * 256) * (long long)HWpix + hw;
    for (int c = 0; c < Cc; ++c) {
        float v = a1[(long long)c * HWpix];
        ob[(long long)c * HWpix] = (half_t)((v - mu) * inv * w[c] + b2[c]);
    }
    for (int c = 0; c < Cc; ++c) {
        float v = a2[(long long)c * HWpix];
        ob[(long long)(Cc + c) * HWpix] = (half_t)((v - mu) * inv * w[Cc + c] + b2[Cc + c]);
    }
}

// depthwise 3x3 central-difference conv: y = dw3x3(x) - theta*x*sum(kernel)
__global__ void dwconv_cd_kernel(const float* __restrict__ x,
                                 const float* __restrict__ wdw,
                                 float* __restrict__ out)
{
    long long i = (long long)blockIdx.x * 256 + threadIdx.x;
    if (i >= (long long)Bn * Cc * HWpix) return;
    int hw = (int)(i & 65535);
    int c  = (int)((i >> 16) & (Cc - 1));
    int h  = hw >> 8, w = hw & 255;
    const float* wc = wdw + c * 9;
    const float* xb = x + (i - hw);
    float acc = 0.f, ks = 0.f;
    for (int dy = -1; dy <= 1; ++dy)
        for (int dx = -1; dx <= 1; ++dx) {
            float wv = wc[(dy + 1) * 3 + (dx + 1)];
            ks += wv;
            int hh = h + dy, ww = w + dx;
            if (hh >= 0 && hh < 256 && ww >= 0 && ww < 256)
                acc += wv * xb[hh * 256 + ww];
        }
    out[i] = acc - THETA * xb[hw] * ks;
}

// chunked deterministic reduction for channel attention:
//   S[c,d] += q_c.k_d ; qq[c] += q_c^2 ; kk[d] += k_d^2   (chunk of 256 n)
__global__ __launch_bounds__(256)
void attn_reduce_kernel(const float* __restrict__ q,
                        const float* __restrict__ k,
                        float* __restrict__ Spart,
                        float* __restrict__ qpart,
                        float* __restrict__ kpart, int nchunks)
{
    __shared__ float qs[16][256];
    __shared__ float ks[16][256];
    int bh = blockIdx.z, chunk = blockIdx.x;
    int b = bh >> 3, hd = bh & 7;
    const float* qb = q + (long long)(b * Cc + hd * 16) * HWpix;
    const float* kb = k + (long long)(b * Cc + hd * 16) * HWpix;
    for (int idx = threadIdx.x; idx < 16 * 256; idx += 256) {
        int c = idx >> 8, i = idx & 255;
        long long n = (long long)chunk * 256 + i;
        qs[c][i] = qb[(long long)c * HWpix + n];
        ks[c][i] = kb[(long long)c * HWpix + n];
    }
    __syncthreads();
    int c = threadIdx.x >> 4, d = threadIdx.x & 15;
    float s = 0.f, qq = 0.f, kk = 0.f;
    for (int i = 0; i < 256; ++i) {
        float qv = qs[c][i], kv = ks[d][i];
        s += qv * kv; qq += qv * qv; kk += kv * kv;
    }
    Spart[((long long)bh * nchunks + chunk) * 256 + threadIdx.x] = s;
    if (d == 0) qpart[((long long)bh * nchunks + chunk) * 16 + c] = qq;
    if (c == 0) kpart[((long long)bh * nchunks + chunk) * 16 + d] = kk;
}

__global__ __launch_bounds__(256)
void attn_softmax_kernel(const float* __restrict__ Spart,
                         const float* __restrict__ qpart,
                         const float* __restrict__ kpart,
                         const float* __restrict__ temp,
                         float* __restrict__ attn, int nchunks)
{
    int bh = blockIdx.x, hd = bh & 7;
    int t = threadIdx.x, c = t >> 4, d = t & 15;
    float S = 0.f, qq = 0.f, kk = 0.f;
    for (int ch = 0; ch < nchunks; ++ch) {
        S  += Spart[((long long)bh * nchunks + ch) * 256 + t];
        qq += qpart[((long long)bh * nchunks + ch) * 16 + c];
        kk += kpart[((long long)bh * nchunks + ch) * 16 + d];
    }
    float qn = fmaxf(sqrtf(qq), 1e-12f);
    float kn = fmaxf(sqrtf(kk), 1e-12f);
    float logit = S / (qn * kn) * temp[hd];
    __shared__ float sh[256];
    sh[t] = logit; __syncthreads();
    float mx = -1e30f;
    for (int j = 0; j < 16; ++j) mx = fmaxf(mx, sh[(c << 4) + j]);
    float e = expf(logit - mx);
    __shared__ float sh2[256];
    sh2[t] = e; __syncthreads();
    float sum = 0.f;
    for (int j = 0; j < 16; ++j) sum += sh2[(c << 4) + j];
    attn[(long long)bh * 256 + t] = e / sum;
}

// out[c,n] = sum_d attn[c,d] * kv[d,n]  (f16 for the proj GEMM)
__global__ __launch_bounds__(256)
void attn_apply_kernel(const float* __restrict__ kv,
                       const float* __restrict__ attn,
                       half_t* __restrict__ out)
{
    __shared__ float a[256];
    int bh = blockIdx.z, b = bh >> 3, hd = bh & 7;
    a[threadIdx.x] = attn[(long long)bh * 256 + threadIdx.x];
    __syncthreads();
    long long n = (long long)blockIdx.x * 256 + threadIdx.x;
    const float* kb = kv + (long long)(b * Cc + hd * 16) * HWpix;
    half_t* ob      = out + (long long)(b * Cc + hd * 16) * HWpix;
    float kvv[16];
    for (int d = 0; d < 16; ++d) kvv[d] = kb[(long long)d * HWpix + n];
    for (int i = 0; i < 16; ++i) {
        float s = 0.f;
        for (int d = 0; d < 16; ++d) s += a[i * 16 + d] * kvv[d];
        ob[(long long)i * HWpix + n] = (half_t)s;
    }
}

// amplitude/phase from complex spectrum, placed at channel offset in concat
// (operates over the padded FLD=136 plane; pad columns carry garbage that
//  never reaches real columns downstream)
__global__ void amppha_kernel(const float* __restrict__ Fre,
                              const float* __restrict__ Fim,
                              half_t* __restrict__ ampcat,
                              half_t* __restrict__ phacat, int chanOff)
{
    long long i = (long long)blockIdx.x * 256 + threadIdx.x;
    if (i >= (long long)Bn * Cc * FPI) return;
    long long bc = i / FPI, f = i % FPI;
    long long b = bc >> 7, c = bc & 127;
    float re = Fre[i], im = Fim[i];
    long long o = ((b * 256) + chanOff + c) * (long long)FPI + f;
    ampcat[o] = (half_t)sqrtf(re * re + im * im);
    phacat[o] = (half_t)atan2f(im, re);
}

// real = amp*cos(pha)+1e-8 ; imag = amp*sin(pha)+1e-8   -> f16 GEMM operands
__global__ void realimag_kernel(const float* __restrict__ amp,
                                const float* __restrict__ pha,
                                half_t* __restrict__ re,
                                half_t* __restrict__ im)
{
    long long i = (long long)blockIdx.x * 256 + threadIdx.x;
    if (i >= (long long)Bn * Cc * FPI) return;
    float a = amp[i], p = pha[i];
    re[i] = (half_t)(a * __cosf(p) + 1e-8f);
    im[i] = (half_t)(a * __sinf(p) + 1e-8f);
}

// gated FFN: out = gelu_exact(x1) * x2  (x1 = rows 0..255, x2 = rows 256..511)
__global__ void gate_kernel(const float* __restrict__ y, half_t* __restrict__ out)
{
    long long i = (long long)blockIdx.x * 256 + threadIdx.x;
    if (i >= (long long)256 * HWpix) return;
    float x1 = y[i], x2 = y[i + (long long)256 * HWpix];
    float g = 0.5f * x1 * (1.0f + erff(x1 * 0.70710678118f));
    out[i] = (half_t)(g * x2);
}

// DFT twiddle tables (f16). Inverse normalization 1/65536 lives in GEMM alpha.
// TWc/TWs use padded leading dim FLD.
__global__ void init_tables_kernel(half_t* TWc, half_t* TWs,
                                   half_t* TCH, half_t* TSH,
                                   half_t* TBC, half_t* TBS)
{
    const float TWO_PI = 6.28318530717958647692f;
    for (long long i = (long long)blockIdx.x * 256 + threadIdx.x;
         i < 256 * 129; i += (long long)gridDim.x * 256) {
        int w = (int)(i / 129), kw = (int)(i % 129);
        float ang = TWO_PI * (float)((w * kw) & 255) / 256.f;
        TWc[w * FLD + kw] = (half_t)cosf(ang);
        TWs[w * FLD + kw] = (half_t)sinf(ang);
    }
    for (long long i = (long long)blockIdx.x * 256 + threadIdx.x;
         i < 256 * 256; i += (long long)gridDim.x * 256) {
        int a = (int)(i >> 8), b = (int)(i & 255);
        float ang = TWO_PI * (float)((a * b) & 255) / 256.f;
        TCH[i] = (half_t)cosf(ang);
        TSH[i] = (half_t)sinf(ang);
    }
    for (long long i = (long long)blockIdx.x * 256 + threadIdx.x;
         i < 129 * 256; i += (long long)gridDim.x * 256) {
        int kw = (int)(i / 256), w = (int)(i % 256);
        float al = (kw == 0 || kw == 128) ? 1.f : 2.f;
        float ang = TWO_PI * (float)((kw * w) & 255) / 256.f;
        TBC[i] = (half_t)(al * cosf(ang));
        TBS[i] = (half_t)(al * sinf(ang));
    }
}

// ===========================================================================
// Host side
// ===========================================================================
static void launch_gemm(hipStream_t st,
                        const half_t* A, long long sA, int lda,
                        const half_t* B, long long sB, int ldb,
                        float* C, long long sC, int ldc,
                        half_t* C16, long long sC16, int ldc16,
                        const float* bias,
                        const float* Res, long long sR, int ldr,
                        int M, int K, int N, float alpha, int accumulate,
                        int act, int batches)
{
    dim3 g((N + TN - 1) / TN, (M + TM - 1) / TM, batches);
    gemm_f16_kernel<<<g, 256, 0, st>>>(A, sA, lda, B, sB, ldb, C, sC, ldc,
                                       C16, sC16, ldc16, bias, Res, sR, ldr,
                                       M, K, N, alpha, accumulate, act);
}

static void launch_cvt(hipStream_t st, const float* in, half_t* out,
                       long long n, int doAbs)
{
    long long blocks = (n + 255) / 256;
    cvt_f32_f16_kernel<<<(unsigned)blocks, 256, 0, st>>>(in, out, n, doAbs);
}

extern "C" void kernel_launch(void* const* d_in, const int* in_sizes, int n_in,
                              void* d_out, int out_size, void* d_ws, size_t ws_size,
                              hipStream_t stream)
{
    (void)in_sizes; (void)n_in; (void)out_size; (void)ws_size;
    // ---- inputs (setup_inputs order, all float32) ----
    const float* ms     = (const float*)d_in[0];
    const float* pan    = (const float*)d_in[1];
    const float* ln1_w  = (const float*)d_in[2];
    const float* ln1_b  = (const float*)d_in[3];
    const float* ln2_w  = (const float*)d_in[4];
    const float* ln2_b  = (const float*)d_in[5];
    const float* qkv_w  = (const float*)d_in[6];
    const float* dw_w   = (const float*)d_in[7];
    const float* temper = (const float*)d_in[8];
    const float* proj_w = (const float*)d_in[9];
    const float* pre1_w = (const float*)d_in[10];
    const float* pre1_b = (const float*)d_in[11];
    const float* pre2_w = (const float*)d_in[12];
    const float* pre2_b = (const float*)d_in[13];
    const float* amp1_w = (const float*)d_in[14];
    const float* amp1_b = (const float*)d_in[15];
    const float* amp2_w = (const float*)d_in[16];
    const float* amp2_b = (const float*)d_in[17];
    const float* pha1_w = (const float*)d_in[18];
    const float* pha1_b = (const float*)d_in[19];
    const float* pha2_w = (const float*)d_in[20];
    const float* pha2_b = (const float*)d_in[21];
    const float* post_w = (const float*)d_in[22];
    const float* post_b = (const float*)d_in[23];
    const float* pin_w  = (const float*)d_in[24];
    const float* pout_w = (const float*)d_in[25];
    float* outp = (float*)d_out;

    // ---- workspace arena ----
    char* ws = (char*)d_ws;
    size_t cur = 0;
    auto alloc = [&](size_t bytes) {
        size_t o = cur; cur = (cur + bytes + 255) & ~(size_t)255; return o;
    };
    // f16 weights
    size_t o_qkv16  = alloc(16384 * 2);
    size_t o_proj16 = alloc(16384 * 2);
    size_t o_pre116 = alloc(16384 * 2);
    size_t o_pre216 = alloc(16384 * 2);
    size_t o_amp116 = alloc(32768 * 2);
    size_t o_amp216 = alloc(16384 * 2);
    size_t o_pha116 = alloc(32768 * 2);
    size_t o_pha216 = alloc(16384 * 2);
    size_t o_post16 = alloc(16384 * 2);
    size_t o_pin16  = alloc(131072 * 2);
    size_t o_pout16 = alloc(32768 * 2);
    // DFT tables (TW* use FLD-padded rows)
    size_t o_TWc = alloc((size_t)256 * FLD * 2), o_TWs = alloc((size_t)256 * FLD * 2);
    size_t o_TCH = alloc(65536 * 2), o_TSH = alloc(65536 * 2);
    size_t o_TBC = alloc(33024 * 2), o_TBS = alloc(33024 * 2);
    // attention stats
    const int NCH = 256; // n-chunks of 256 -> 65536
    size_t o_Spart = alloc((size_t)32 * NCH * 256 * 4);
    size_t o_qpart = alloc((size_t)32 * NCH * 16 * 4);
    size_t o_kpart = alloc((size_t)32 * NCH * 16 * 4);
    size_t o_attn  = alloc((size_t)32 * 256 * 4);

    // big slots (manually scheduled lifetimes), 136 MiB each
    const size_t SLOT = (size_t)136 << 20;
    size_t bigBase = (cur + 4095) & ~(size_t)4095;
    char* P0 = ws + bigBase + 0 * SLOT;  // x (residual stream), persists
    char* P1 = ws + bigBase + 1 * SLOT;  // phacat16 then frefuse
    char* S1 = ws + bigBase + 2 * SLOT;
    char* S2 = ws + bigBase + 3 * SLOT;
    char* S3 = ws + bigBase + 4 * SLOT;
    char* S4 = ws + bigBase + 5 * SLOT;
    char* S5 = ws + bigBase + 6 * SLOT;
    char* S6 = ws + bigBase + 7 * SLOT;

    const long long CHW  = (long long)Cc * HWpix;       // per-batch C*HW elems
    const long long FQP  = (long long)Bn * Cc * FPI;    // padded freq elems
    const size_t CHW16b = (size_t)CHW * Bn * 2;         // f16 bytes, all batches
    const size_t FQ16b  = (size_t)FQP * 2;              // 35.7 MB
    const size_t FQ32b  = (size_t)FQP * 4;              // 142.6/2 MB

    // ---- 0) tables + f16 weights ----
    init_tables_kernel<<<256, 256, 0, stream>>>(
        (half_t*)(ws + o_TWc), (half_t*)(ws + o_TWs),
        (half_t*)(ws + o_TCH), (half_t*)(ws + o_TSH),
        (half_t*)(ws + o_TBC), (half_t*)(ws + o_TBS));
    launch_cvt(stream, qkv_w,  (half_t*)(ws + o_qkv16),  16384, 0);
    launch_cvt(stream, proj_w, (half_t*)(ws + o_proj16), 16384, 0);
    launch_cvt(stream, pre1_w, (half_t*)(ws + o_pre116), 16384, 0);
    launch_cvt(stream, pre2_w, (half_t*)(ws + o_pre216), 16384, 0);
    launch_cvt(stream, amp1_w, (half_t*)(ws + o_amp116), 32768, 0);
    launch_cvt(stream, amp2_w, (half_t*)(ws + o_amp216), 16384, 0);
    launch_cvt(stream, pha1_w, (half_t*)(ws + o_pha116), 32768, 0);
    launch_cvt(stream, pha2_w, (half_t*)(ws + o_pha216), 16384, 0);
    launch_cvt(stream, post_w, (half_t*)(ws + o_post16), 16384, 0);
    launch_cvt(stream, pin_w,  (half_t*)(ws + o_pin16),  131072, 0);
    launch_cvt(stream, pout_w, (half_t*)(ws + o_pout16), 32768, 0);

    // ======================= Attention branch =======================
    half_t* lnms16 = (half_t*)S1;
    half_t* lnpan16 = (half_t*)(S1 + CHW16b);
    const unsigned lnBlocks = (unsigned)((Bn * (long long)HWpix + 255) / 256);
    layernorm_kernel<<<lnBlocks, 256, 0, stream>>>(ms,  ln1_w, ln1_b, lnms16);
    layernorm_kernel<<<lnBlocks, 256, 0, stream>>>(pan, ln1_w, ln1_b, lnpan16);

    float* q0  = (float*)S2;
    float* kv0 = (float*)S3;
    launch_gemm(stream, (half_t*)(ws + o_qkv16), 0, Cc, lnms16, CHW, HWpix,
                q0, CHW, HWpix, nullptr, 0, 0, nullptr, nullptr, 0, 0,
                Cc, Cc, HWpix, 1.f, 0, 0, Bn);
    launch_gemm(stream, (half_t*)(ws + o_qkv16), 0, Cc, lnpan16, CHW, HWpix,
                kv0, CHW, HWpix, nullptr, 0, 0, nullptr, nullptr, 0, 0,
                Cc, Cc, HWpix, 1.f, 0, 0, Bn);

    float* qf  = (float*)S4;
    float* kvf = (float*)S5;
    const unsigned dwBlocks = (unsigned)(((long long)Bn * CHW + 255) / 256);
    dwconv_cd_kernel<<<dwBlocks, 256, 0, stream>>>(q0,  dw_w, qf);
    dwconv_cd_kernel<<<dwBlocks, 256, 0, stream>>>(kv0, dw_w, kvf);

    attn_reduce_kernel<<<dim3(NCH, 1, 32), 256, 0, stream>>>(
        qf, kvf, (float*)(ws + o_Spart), (float*)(ws + o_qpart),
        (float*)(ws + o_kpart), NCH);
    attn_softmax_kernel<<<32, 256, 0, stream>>>(
        (float*)(ws + o_Spart), (float*)(ws + o_qpart),
        (float*)(ws + o_kpart), temper, (float*)(ws + o_attn), NCH);

    half_t* attnout16 = (half_t*)S2;   // q0 dead
    attn_apply_kernel<<<dim3(HWpix / 256, 1, 32), 256, 0, stream>>>(
        kvf, (float*)(ws + o_attn), attnout16);

    float* xf = (float*)P0;            // x = ms + proj(attn_out)
    launch_gemm(stream, (half_t*)(ws + o_proj16), 0, Cc, attnout16, CHW, HWpix,
                xf, CHW, HWpix, nullptr, 0, 0, nullptr, ms, CHW, HWpix,
                Cc, Cc, HWpix, 1.f, 0, 0, Bn);

    // ======================= Frequency branch =======================
    half_t* ms16  = (half_t*)S1;       // lnms dead
    half_t* pan16 = (half_t*)(S1 + CHW16b);
    launch_cvt(stream, ms,  ms16,  (long long)Bn * CHW, 0);
    launch_cvt(stream, pan, pan16, (long long)Bn * CHW, 0);

    half_t* ampcat16 = (half_t*)S5;    // kv dead after apply
    half_t* phacat16 = (half_t*)P1;
    half_t* pre16  = (half_t*)S2;
    half_t* R1_16  = (half_t*)(S2 + CHW16b);
    half_t* I1_16  = (half_t*)(S2 + CHW16b + FQ16b / 2);
    float*  Fre    = (float*)S3;
    float*  Fim    = (float*)S4;       // q dead after reduce

    const unsigned fqBlocks = (unsigned)((FQP + 255) / 256);
    for (int t = 0; t < 2; ++t) {
        const half_t* src16 = t == 0 ? ms16 : pan16;
        const half_t* prew  = (half_t*)(ws + (t == 0 ? o_pre116 : o_pre216));
        const float*  preb  = t == 0 ? pre1_b : pre2_b;
        // pre conv (f16 shadow only)
        launch_gemm(stream, prew, 0, Cc, src16, CHW, HWpix,
                    nullptr, 0, 0, pre16, CHW, HWpix, preb,
                    nullptr, 0, 0, Cc, Cc, HWpix, 1.f, 0, 0, Bn);
        // stage 1: row DFT  (A = image rows, B = W-twiddles)
        launch_gemm(stream, pre16, 65536, 256, (half_t*)(ws + o_TWc), 0, FLD,
                    nullptr, 0, 0, R1_16, FPI, FLD, nullptr, nullptr, 0, 0,
                    256, 256, 129, 1.f, 0, 0, Bn * Cc);
        launch_gemm(stream, pre16, 65536, 256, (half_t*)(ws + o_TWs), 0, FLD,
                    nullptr, 0, 0, I1_16, FPI, FLD, nullptr, nullptr, 0, 0,
                    256, 256, 129, -1.f, 0, 0, Bn * Cc);
        // stage 2: column DFT  Fre = CH*R1 + SH*I1 ; Fim = CH*I1 - SH*R1
        launch_gemm(stream, (half_t*)(ws + o_TCH), 0, 256, R1_16, FPI, FLD,
                    Fre, FPI, FLD, nullptr, 0, 0, nullptr, nullptr, 0, 0,
                    256, 256, 129, 1.f, 0, 0, Bn * Cc);
        launch_gemm(stream, (half_t*)(ws + o_TSH), 0, 256, I1_16, FPI, FLD,
                    Fre, FPI, FLD, nullptr, 0, 0, nullptr, nullptr, 0, 0,
                    256, 256, 129, 1.f, 1, 0, Bn * Cc);
        launch_gemm(stream, (half_t*)(ws + o_TCH), 0, 256, I1_16, FPI, FLD,
                    Fim, FPI, FLD, nullptr, 0, 0, nullptr, nullptr, 0, 0,
                    256, 256, 129, 1.f, 0, 0, Bn * Cc);
        launch_gemm(stream, (half_t*)(ws + o_TSH), 0, 256, R1_16, FPI, FLD,
                    Fim, FPI, FLD, nullptr, 0, 0, nullptr, nullptr, 0, 0,
                    256, 256, 129, -1.f, 1, 0, Bn * Cc);
        amppha_kernel<<<fqBlocks, 256, 0, stream>>>(Fre, Fim, ampcat16,
                                                    phacat16, t * Cc);
    }

    // amp / pha MLPs over 2C channels (per batch, N = FPI, 100% fast path)
    half_t* mid16  = (half_t*)S2;                 // pre/R1/I1 dead
    float*  ampout = (float*)(S2 + FQ16b);
    launch_gemm(stream, (half_t*)(ws + o_amp116), 0, 256, ampcat16,
                (long long)256 * FPI, FPI, nullptr, 0, 0,
                mid16, (long long)Cc * FPI, FPI, amp1_b, nullptr, 0, 0,
                Cc, 256, FPI, 1.f, 0, 1, Bn);
    launch_gemm(stream, (half_t*)(ws + o_amp216), 0, Cc, mid16,
                (long long)Cc * FPI, FPI, ampout, (long long)Cc * FPI,
                FPI, nullptr, 0, 0, amp2_b, nullptr, 0, 0,
                Cc, Cc, FPI, 1.f, 0, 0, Bn);
    half_t* pmid16 = (half_t*)S4;                 // Fim dead
    float*  phaout = (float*)(S4 + FQ16b);
    launch_gemm(stream, (half_t*)(ws + o_pha116), 0, 256, phacat16,
                (long long)256 * FPI, FPI, nullptr, 0, 0,
                pmid16, (long long)Cc * FPI, FPI, pha1_b, nullptr, 0, 0,
                Cc, 256, FPI, 1.f, 0, 1, Bn);
    launch_gemm(stream, (half_t*)(ws + o_pha216), 0, Cc, pmid16,
                (long long)Cc * FPI, FPI, phaout, (long long)Cc * FPI,
                FPI, nullptr, 0, 0, pha2_b, nullptr, 0, 0,
                Cc, Cc, FPI, 1.f, 0, 0, Bn);

    // back to complex, then inverse transform (DFT-by-GEMM)
    half_t* re16 = (half_t*)S3;                   // Fre dead
    half_t* im16 = (half_t*)(S3 + FQ16b);
    realimag_kernel<<<fqBlocks, 256, 0, stream>>>(ampout, phaout, re16, im16);

    float* Gre = (float*)S4;                      // pmid/phaout consumed
    float* Gim = (float*)(S4 + FQ32b);
    launch_gemm(stream, (half_t*)(ws + o_TCH), 0, 256, re16, FPI, FLD,
                Gre, FPI, FLD, nullptr, 0, 0, nullptr, nullptr, 0, 0,
                256, 256, 129, 1.f, 0, 0, Bn * Cc);
    launch_gemm(stream, (half_t*)(ws + o_TSH), 0, 256, im16, FPI, FLD,
                Gre, FPI, FLD, nullptr, 0, 0, nullptr, nullptr, 0, 0,
                256, 256, 129, -1.f, 1, 0, Bn * Cc);
    launch_gemm(stream, (half_t*)(ws + o_TCH), 0, 256, im16, FPI, FLD,
                Gim, FPI, FLD, nullptr, 0, 0, nullptr, nullptr, 0, 0,
                256, 256, 129, 1.f, 0, 0, Bn * Cc);
    launch_gemm(stream, (half_t*)(ws + o_TSH), 0, 256, re16, FPI, FLD,
                Gim, FPI, FLD, nullptr, 0, 0, nullptr, nullptr, 0, 0,
                256, 256, 129, 1.f, 1, 0, Bn * Cc);

    half_t* Gre16 = (half_t*)S2;                  // amp chain dead
    half_t* Gim16 = (half_t*)(S2 + FQ16b);
    launch_cvt(stream, Gre, Gre16, FQP, 0);
    launch_cvt(stream, Gim, Gim16, FQP, 0);

    float* xir = (float*)S5;                      // ampcat dead
    const float INVN = 1.0f / 65536.0f;
    launch_gemm(stream, Gre16, FPI, FLD, (half_t*)(ws + o_TBC), 0, 256,
                xir, HWpix, 256, nullptr, 0, 0, nullptr, nullptr, 0, 0,
                256, 129, 256, INVN, 0, 0, Bn * Cc);
    launch_gemm(stream, Gim16, FPI, FLD, (half_t*)(ws + o_TBS), 0, 256,
                xir, HWpix, 256, nullptr, 0, 0, nullptr, nullptr, 0, 0,
                256, 129, 256, -INVN, 1, 0, Bn * Cc);

    half_t* xabs16 = (half_t*)S6;
    launch_cvt(stream, xir, xabs16, (long long)Bn * CHW, /*abs*/1);

    float* frefuse = (float*)P1;                  // phacat dead
    launch_gemm(stream, (half_t*)(ws + o_post16), 0, Cc, xabs16, CHW, HWpix,
                frefuse, CHW, HWpix, nullptr, 0, 0, post_b, nullptr, 0, 0,
                Cc, Cc, HWpix, 1.f, 0, 0, Bn);

    // ======================= Gated FFN =======================
    half_t* ln2cat16 = (half_t*)S1;               // ms16/pan16 dead
    layernorm2_cat_kernel<<<lnBlocks, 256, 0, stream>>>(xf, frefuse, ln2_w,
                                                        ln2_b, ln2cat16);
    float*  ybuf    = (float*)S2;                 // Gre16/Gim16 dead
    half_t* gated16 = (half_t*)S3;                // re16/im16 dead
    const unsigned gateBlocks = (unsigned)(((long long)256 * HWpix + 255) / 256);
    for (int b = 0; b < Bn; ++b) {
        launch_gemm(stream, (half_t*)(ws + o_pin16), 0, 256,
                    ln2cat16 + (long long)b * 256 * HWpix, 0, HWpix,
                    ybuf, 0, HWpix, nullptr, 0, 0, nullptr, nullptr, 0, 0,
                    512, 256, HWpix, 1.f, 0, 0, 1);
        gate_kernel<<<gateBlocks, 256, 0, stream>>>(ybuf, gated16);
        launch_gemm(stream, (half_t*)(ws + o_pout16), 0, 256,
                    gated16, 0, HWpix,
                    outp + (long long)b * CHW, 0, HWpix, nullptr, 0, 0,
                    nullptr, xf + (long long)b * CHW, 0, HWpix,
                    Cc, 256, HWpix, 1.f, 0, 0, 1);
    }
}
